// MultiHeadAttention_77360950936277
// MI455X (gfx1250) — matrix operand order repaired and checked by reading, not compiled, not run on hardware
//
#include <hip/hip_runtime.h>
#include <hip/hip_bf16.h>
#include <math.h>

typedef _Float16 half8 __attribute__((ext_vector_type(8)));
typedef _Float16 v16h  __attribute__((ext_vector_type(16)));
typedef float    v8f   __attribute__((ext_vector_type(8)));

#define H_    4
#define B_    8
#define T_    2048
#define DH_   64
#define DMID_ 256
#define ROWS_ (B_ * T_)          // 16384

union AFrag { v16h v; half8 h[2]; };
union BFrag { v16h v; half8 h[2]; unsigned int u[8]; };
union HPack { unsigned int u; _Float16 e[2]; };

// ---------------------------------------------------------------------------
// Kernel 1: QKV projection. X:[ROWS,64] f32, W*:[64,256] f32.
// Q is pre-scaled by 1/64 (exact power of two in f16).
// Qh,Kh: [H*B][T][64] f16 row-major.  Vt: [H*B][64][T] f16 (transposed).
// ---------------------------------------------------------------------------
__global__ void __launch_bounds__(256)
qkv_proj(const float* __restrict__ X,
         const float* __restrict__ W1, const float* __restrict__ W2,
         const float* __restrict__ W3,
         _Float16* __restrict__ Qh, _Float16* __restrict__ Kh,
         _Float16* __restrict__ Vt) {
  __shared__ float xrow[DH_];
  const int row = blockIdx.x;            // 0..ROWS_-1
  const int b   = row >> 11;             // row / T_
  const int t   = row & (T_ - 1);
  const int c   = threadIdx.x;           // output column 0..255
  if (c < DH_) xrow[c] = X[(size_t)row * DH_ + c];
  __syncthreads();

  float q = 0.f, k = 0.f, v = 0.f;
#pragma unroll 8
  for (int i = 0; i < DH_; ++i) {
    const float x = xrow[i];
    q = fmaf(x, W1[i * DMID_ + c], q);
    k = fmaf(x, W2[i * DMID_ + c], k);
    v = fmaf(x, W3[i * DMID_ + c], v);
  }
  const int h = c >> 6, d = c & 63;
  const size_t hb = (size_t)(h * B_ + b);
  Qh[(hb * T_ + t) * DH_ + d] = (_Float16)(q * 0.015625f);  // fold 1/64 scale
  Kh[(hb * T_ + t) * DH_ + d] = (_Float16)k;
  Vt[(hb * DH_ + d) * T_ + t] = (_Float16)v;
}

// ---------------------------------------------------------------------------
// Kernel 2: flash attention, transposed formulation (lane == query column).
//   S^T  = K · Q^T          (A = K rows, B = Q rows)
//   ctx^T = V^T · P^T       (A = Vt rows, B = P^T via 4 packed shfl_xor)
// Grid: H*B*(T/128) blocks, 256 threads (8 waves); wave -> 16-query tile.
// gpart: [H][ROWS] f32 (each element written exactly once -> deterministic).
// ---------------------------------------------------------------------------
__global__ void __launch_bounds__(256)
flash_attn(const _Float16* __restrict__ Qh, const _Float16* __restrict__ Kh,
           const _Float16* __restrict__ Vt, const float* __restrict__ Wout,
           float* __restrict__ gpart) {
  const int lane = threadIdx.x & 31;
  const int wave = threadIdx.x >> 5;
  const int n    = lane & 15;     // query column owned by this lane
  const int hi   = lane >> 4;     // half-group selector

  const int blk  = blockIdx.x;
  const int qblk = blk & 15;      // T_/128 = 16 q-blocks per (h,b)
  const int hb   = blk >> 4;      // h*B + b
  const int h    = hb >> 3;
  const int b    = hb & 7;

  const _Float16* Qp = Qh + (size_t)hb * T_ * DH_;
  const _Float16* Kp = Kh + (size_t)hb * T_ * DH_;
  const _Float16* Vp = Vt + (size_t)hb * DH_ * T_;
  const int q0 = qblk * 128 + wave * 16;   // first query of this wave's tile

  // ---- Q as B-operand of K*Q^T: lane = query column n, halves = d ----
  BFrag qb0, qb1;
  {
    const half8* qrow = (const half8*)(Qp + (size_t)(q0 + n) * DH_);
    qb0.h[0] = qrow[hi];             // d = hi*8 + 0..7
    qb0.h[1] = qrow[2 + hi];         // d = 16 + hi*8 + 0..7
    qb1.h[0] = qrow[4 + hi];         // d = 32 + hi*8 + 0..7
    qb1.h[1] = qrow[6 + hi];
  }

  v8f acc[4] = {};      // ctx^T accumulator: d = dt*16 + v + 8*hi, col = query n
  float m = -INFINITY;  // running max for query n (replicated in both hi lanes)
  float l = 0.f;        // running denom for query n

  for (int kb = 0; kb < T_; kb += 32) {
    // ---- S^T for two 16-key subtiles: rows = keys, cols = queries ----
    v8f st0 = {}, st1 = {};
    {
      const half8* kr = (const half8*)(Kp + (size_t)(kb + n) * DH_);
      AFrag a0, a1;
      a0.h[0] = kr[hi];         // d = hi*8 + 0..7
      a0.h[1] = kr[2 + hi];     // d = 16 + hi*8 + 0..7
      a1.h[0] = kr[4 + hi];     // d = 32 + hi*8 + 0..7
      a1.h[1] = kr[6 + hi];     // d = 48 + hi*8 + 0..7
      st0 = __builtin_amdgcn_wmma_f32_16x16x32_f16(false, a0.v, false, qb0.v,
                                                   (short)0, st0, false, false);
      st0 = __builtin_amdgcn_wmma_f32_16x16x32_f16(false, a1.v, false, qb1.v,
                                                   (short)0, st0, false, false);
    }
    {
      const half8* kr = (const half8*)(Kp + (size_t)(kb + 16 + n) * DH_);
      AFrag a0, a1;
      a0.h[0] = kr[hi];
      a0.h[1] = kr[2 + hi];
      a1.h[0] = kr[4 + hi];
      a1.h[1] = kr[6 + hi];
      st1 = __builtin_amdgcn_wmma_f32_16x16x32_f16(false, a0.v, false, qb0.v,
                                                   (short)0, st1, false, false);
      st1 = __builtin_amdgcn_wmma_f32_16x16x32_f16(false, a1.v, false, qb1.v,
                                                   (short)0, st1, false, false);
    }

    // ---- online softmax per query column: in-lane + ONE shfl_xor(16) ----
    float mloc = fmaxf(st0[0], st1[0]);
#pragma unroll
    for (int v = 1; v < 8; ++v) mloc = fmaxf(mloc, fmaxf(st0[v], st1[v]));
    mloc = fmaxf(mloc, __shfl_xor(mloc, 16));
    const float mn = fmaxf(m, mloc);
    const float cf = __expf(m - mn);     // exp(-inf)=0 on first iteration
    m = mn;

    float ps0[8], ps1[8];
    float ssum = 0.f;
#pragma unroll
    for (int v = 0; v < 8; ++v) {
      const float e0 = __expf(st0[v] - mn);
      const float e1 = __expf(st1[v] - mn);
      ps0[v] = e0; ps1[v] = e1;
      ssum += e0 + e1;
    }
    ssum += __shfl_xor(ssum, 16);
    l = l * cf + ssum;
#pragma unroll
    for (int dt = 0; dt < 4; ++dt)
#pragma unroll
      for (int v = 0; v < 8; ++v) acc[dt][v] *= cf;

    // ---- build P^T B-fragment: lane (n,hi) needs subtile 'hi' from both
    //      hi-groups. keep = own subtile-hi values (halves v+8*hi);
    //      recv = partner's subtile-hi values (halves v+8*(1-hi)).
    //      Packed as f16 pairs -> 4 shfl_xor total. ----
    BFrag pb;
#pragma unroll
    for (int w = 0; w < 4; ++w) {
      HPack kp, sp, rp;
      const float k0 = hi ? ps1[2 * w]     : ps0[2 * w];
      const float k1 = hi ? ps1[2 * w + 1] : ps0[2 * w + 1];
      const float s0 = hi ? ps0[2 * w]     : ps1[2 * w];
      const float s1 = hi ? ps0[2 * w + 1] : ps1[2 * w + 1];
      kp.e[0] = (_Float16)k0; kp.e[1] = (_Float16)k1;
      sp.e[0] = (_Float16)s0; sp.e[1] = (_Float16)s1;
      rp.u = (unsigned int)__shfl_xor((int)sp.u, 16);
      pb.u[4 * hi + w]       = kp.u;   // own half-region
      pb.u[4 * (1 - hi) + w] = rp.u;   // partner half-region
    }

    // ---- ctx^T += V^T * P^T  (A = Vt rows: lane row = d, halves = keys) ----
#pragma unroll
    for (int dt = 0; dt < 4; ++dt) {
      const half8* vr = (const half8*)(Vp + (size_t)(dt * 16 + n) * T_ + kb);
      AFrag va;
      va.h[0] = vr[2 * hi];     // key = kb + hi*16 + 0..7
      va.h[1] = vr[2 * hi + 1]; // key = kb + hi*16 + 8..15
      acc[dt] = __builtin_amdgcn_wmma_f32_16x16x32_f16(false, va.v, false, pb.v,
                                                       (short)0, acc[dt],
                                                       false, false);
    }
  }

  // ---- gate partial for query n: sum_d ctx[q][d]*Wout[h*64+d], then /l ----
  const float* Wo = Wout + h * DH_;
  float g = 0.f;
#pragma unroll
  for (int dt = 0; dt < 4; ++dt)
#pragma unroll
    for (int v = 0; v < 8; ++v)
      g += acc[dt][v] * Wo[dt * 16 + v + 8 * hi];
  g += __shfl_xor(g, 16);
  g /= l;
  if (hi == 0) {
    const int row = b * T_ + q0 + n;
    gpart[(size_t)h * ROWS_ + row] = g;
  }
}

// ---------------------------------------------------------------------------
// Kernel 3: gate = sum_h gpart[h][row];  out = x * gate
// ---------------------------------------------------------------------------
__global__ void __launch_bounds__(256)
gate_mul(const float* __restrict__ X, const float* __restrict__ gpart,
         float* __restrict__ out) {
  const size_t idx = (size_t)blockIdx.x * blockDim.x + threadIdx.x;
  const size_t row = idx >> 6;
  const float g = gpart[row] + gpart[(size_t)ROWS_ + row] +
                  gpart[2 * (size_t)ROWS_ + row] + gpart[3 * (size_t)ROWS_ + row];
  out[idx] = X[idx] * g;
}

// ---------------------------------------------------------------------------
extern "C" void kernel_launch(void* const* d_in, const int* in_sizes, int n_in,
                              void* d_out, int out_size, void* d_ws, size_t ws_size,
                              hipStream_t stream) {
  const float* X    = (const float*)d_in[0];
  const float* W1   = (const float*)d_in[1];
  const float* W2   = (const float*)d_in[2];
  const float* W3   = (const float*)d_in[3];
  const float* Wout = (const float*)d_in[4];
  float* out = (float*)d_out;

  char* ws = (char*)d_ws;
  const size_t QKV_BYTES = (size_t)H_ * B_ * T_ * DH_ * sizeof(_Float16); // 8 MB
  _Float16* Qh = (_Float16*)(ws);
  _Float16* Kh = (_Float16*)(ws + QKV_BYTES);
  _Float16* Vt = (_Float16*)(ws + 2 * QKV_BYTES);
  float*  gpart = (float*)(ws + 3 * QKV_BYTES);                           // 256 KB

  (void)in_sizes; (void)n_in; (void)out_size; (void)ws_size;

  qkv_proj<<<ROWS_, 256, 0, stream>>>(X, W1, W2, W3, Qh, Kh, Vt);
  flash_attn<<<H_ * B_ * (T_ / 128), 256, 0, stream>>>(Qh, Kh, Vt, Wout, gpart);
  gate_mul<<<(ROWS_ * DH_) / 256, 256, 0, stream>>>(X, gpart, out);
}
